// MultiBranchWithGlobalMamba_62448824484302
// MI455X (gfx1250) — matrix operand order repaired and checked by reading, not compiled, not run on hardware
//
#include <hip/hip_runtime.h>
#include <hip/hip_bf16.h>

typedef __attribute__((ext_vector_type(16))) _Float16 v16h;
typedef __attribute__((ext_vector_type(8)))  float    v8f;

#define ROWS 16          // rows per wave tile
#define WAVES 2          // waves (independent tiles) per workgroup
#define BLOCK (WAVES*32)

// ---------------- per-wave LDS slab ----------------
struct alignas(32) WaveLds {
    float xbuf[16][32];      // input features (12 cols, zero padded to 32)
    float xpad[16];          // zeroed guard so shifted fragA reads stay finite
    float hid [16][64];      // MLP hidden / head hidden
    float seqy[16][192];     // 6 x 32 branch embeddings -> later mamba y_l (head input)
    float gxl [16][32];      // current 32-wide token (embed / gembed output)
    float ring[4][16][64];   // causal-conv ring of xi
    float zb  [16][64];      // z gate
    float xc  [16][64];      // conv+silu output
    float dblv[16][48];      // xproj output (34 valid cols, ld 48)
    float yb  [16][64];      // scan output
    float bc  [16];          // rain: dot(B,C) per row
    float h   [16][64][16];  // mamba state
};

// ---------------- shared staged weights (f16, padded, B-fragment friendly) ----------------
struct alignas(32) SharedW {
    _Float16 gembed[32][32];     // gembed_w
    _Float16 g_in[128][32];      // gmamba in_w (rows 0..63 xi, 64..127 z)
    _Float16 g_xproj[48][64];    // gmamba xproj_w, rows 34..47 zero
    _Float16 g_out[32][64];      // gmamba out_w
    _Float16 w1p[6][64][32];     // slots 0..4: branch w1 (K padded to 32); slot 5: rain embed (rows 32..63 zero)
};

__device__ __forceinline__ float siluf(float v) { return v * (1.f / (1.f + __expf(-v))); }
__device__ __forceinline__ float softplusf(float v) { return (v > 20.f) ? v : __logf(1.f + __expf(v)); }

// A fragment: 16x32 f16 tile from f32 LDS activations (4x ds_load_b128 + cvt)
__device__ __forceinline__ v16h fragA(const float* base, int ldk, int ks) {
    const int l   = threadIdx.x & 31;
    const int row = l & 15;
    const int kb  = ks * 32 + ((l >> 4) << 4);
    const float4* p = reinterpret_cast<const float4*>(base + row * ldk + kb);
    const float4 x0 = p[0], x1 = p[1], x2 = p[2], x3 = p[3];
    v16h a;
    a[0]=(_Float16)x0.x; a[1]=(_Float16)x0.y; a[2]=(_Float16)x0.z; a[3]=(_Float16)x0.w;
    a[4]=(_Float16)x1.x; a[5]=(_Float16)x1.y; a[6]=(_Float16)x1.z; a[7]=(_Float16)x1.w;
    a[8]=(_Float16)x2.x; a[9]=(_Float16)x2.y; a[10]=(_Float16)x2.z; a[11]=(_Float16)x2.w;
    a[12]=(_Float16)x3.x; a[13]=(_Float16)x3.y; a[14]=(_Float16)x3.z; a[15]=(_Float16)x3.w;
    return a;
}

// B fragment from staged f16 LDS weights: one aligned 32B vector load (2x ds_load_b128)
__device__ __forceinline__ v16h fragB_lds(const _Float16* Wl, int ldk, int nt, int ks) {
    const int l  = threadIdx.x & 31;
    const int n  = nt * 16 + (l & 15);
    const int kb = ks * 32 + ((l >> 4) << 4);
    return *reinterpret_cast<const v16h*>(Wl + n * ldk + kb);
}

// B fragment straight from global f32 W[N][K], K multiple of 32 (4x global_load_b128), masked ragged N
__device__ __forceinline__ v16h fragB32(const float* W, int N, int K, int nt, int ks) {
    const int l  = threadIdx.x & 31;
    const int n  = nt * 16 + (l & 15);
    const int kb = ks * 32 + ((l >> 4) << 4);
    const float m = (n < N) ? 1.f : 0.f;
    const float4* p = reinterpret_cast<const float4*>(W + (size_t)(n < N ? n : 0) * K + kb);
    const float4 q0 = p[0], q1 = p[1], q2 = p[2], q3 = p[3];
    v16h b;
    b[0]=(_Float16)(q0.x*m); b[1]=(_Float16)(q0.y*m); b[2]=(_Float16)(q0.z*m); b[3]=(_Float16)(q0.w*m);
    b[4]=(_Float16)(q1.x*m); b[5]=(_Float16)(q1.y*m); b[6]=(_Float16)(q1.z*m); b[7]=(_Float16)(q1.w*m);
    b[8]=(_Float16)(q2.x*m); b[9]=(_Float16)(q2.y*m); b[10]=(_Float16)(q2.z*m); b[11]=(_Float16)(q2.w*m);
    b[12]=(_Float16)(q3.x*m); b[13]=(_Float16)(q3.y*m); b[14]=(_Float16)(q3.z*m); b[15]=(_Float16)(q3.w*m);
    return b;
}

// LDS-weight GEMM: dst[16][N(ldn)] = act(A[16][K] @ Wl^T + bias); N mult of 16, K mult of 32
template<bool HB, bool RELU>
__device__ __forceinline__ void gemmL(float* dst, int ldn, const float* A, int ldk,
                                      const _Float16* Wl, int wldk, const float* bias,
                                      int N, int K) {
    const int ksteps = K >> 5, ntiles = N >> 4;
#pragma unroll
    for (int nt = 0; nt < ntiles; ++nt) {
        v8f c = {};
#pragma unroll
        for (int ks = 0; ks < ksteps; ++ks)
            c = __builtin_amdgcn_wmma_f32_16x16x32_f16(false, fragA(A, ldk, ks),
                                                       false, fragB_lds(Wl, wldk, nt, ks),
                                                       (short)0, c, false, false);
        const int l = threadIdx.x & 31, nc = (l & 15) + nt * 16, mb = (l >> 4) << 3;
        const float bv = HB ? bias[nc] : 0.f;
#pragma unroll
        for (int v = 0; v < 8; ++v) {
            float r = c[v] + bv;
            if (RELU) r = fmaxf(r, 0.f);
            dst[(mb + v) * ldn + nc] = r;
        }
    }
}

// Global-weight GEMM (single-use weights); handles ragged N via mask in fragB32 + store guard
template<bool HB, bool RELU>
__device__ __forceinline__ void gemmG(float* dst, int ldn, const float* A, int ldk,
                                      const float* W, const float* bias, int N, int K) {
    const int ksteps = K >> 5, ntiles = (N + 15) >> 4;
#pragma unroll
    for (int nt = 0; nt < ntiles; ++nt) {
        v8f c = {};
#pragma unroll
        for (int ks = 0; ks < ksteps; ++ks)
            c = __builtin_amdgcn_wmma_f32_16x16x32_f16(false, fragA(A, ldk, ks),
                                                       false, fragB32(W, N, K, nt, ks),
                                                       (short)0, c, false, false);
        const int l = threadIdx.x & 31, nc = (l & 15) + nt * 16, mb = (l >> 4) << 3;
        if (nc < N) {
            const float bv = HB ? bias[nc] : 0.f;
#pragma unroll
            for (int v = 0; v < 8; ++v) {
                float r = c[v] + bv;
                if (RELU) r = fmaxf(r, 0.f);
                dst[(mb + v) * ldn + nc] = r;
            }
        }
    }
}

struct KParams {
    const float* x;
    const float* mw1[5]; const float* mb1[5]; const float* mw2[5]; const float* mb2[5];
    const float* r_embed_w; const float* r_embed_b;
    const float* r_in_w; const float* r_conv_w; const float* r_conv_b;
    const float* r_xproj_w; const float* r_dt_w; const float* r_dt_b;
    const float* r_D; const float* r_out_w;
    const float* gembed_w; const float* gembed_b;
    const float* g_in_w; const float* g_conv_w; const float* g_conv_b;
    const float* g_xproj_w; const float* g_dt_w; const float* g_dt_b;
    const float* g_Alog; const float* g_D; const float* g_out_w;
    const float* head_w1; const float* head_b1; const float* head_w2; const float* head_b2;
    float* out;
};

__global__ __launch_bounds__(BLOCK, 1)
void fused_multibranch_mamba(KParams P) {
    __shared__ WaveLds lds[WAVES];
    __shared__ SharedW sw;
    __shared__ float negA[64][16];

    const int lane = threadIdx.x & 31;
    const int wv   = threadIdx.x >> 5;
    const int row0 = (blockIdx.x * WAVES + wv) * ROWS;
    WaveLds& L = lds[wv];

    const int nfs[5] = {2, 2, 1, 1, 2};

    // ---- one-time staging: negA and f16 padded weights (block-wide, then barrier) ----
    for (int i = threadIdx.x; i < 64 * 16; i += BLOCK)
        (&negA[0][0])[i] = -__expf(P.g_Alog[i]);
    for (int i = threadIdx.x; i < 32 * 32; i += BLOCK)
        (&sw.gembed[0][0])[i] = (_Float16)P.gembed_w[i];
    for (int i = threadIdx.x; i < 128 * 32; i += BLOCK)
        (&sw.g_in[0][0])[i] = (_Float16)P.g_in_w[i];
    for (int i = threadIdx.x; i < 48 * 64; i += BLOCK) {
        const int r = i >> 6, k = i & 63;
        (&sw.g_xproj[0][0])[i] = (_Float16)(r < 34 ? P.g_xproj_w[r * 64 + k] : 0.f);
    }
    for (int i = threadIdx.x; i < 32 * 64; i += BLOCK)
        (&sw.g_out[0][0])[i] = (_Float16)P.g_out_w[i];
#pragma unroll
    for (int g = 0; g < 5; ++g) {
        const int nf = nfs[g];
        for (int i = threadIdx.x; i < 64 * 32; i += BLOCK) {
            const int r = i >> 5, k = i & 31;
            (&sw.w1p[g][0][0])[i] = (_Float16)(k < nf ? P.mw1[g][r * nf + k] : 0.f);
        }
    }
    for (int i = threadIdx.x; i < 64 * 32; i += BLOCK) {
        const int r = i >> 5, k = i & 31;
        (&sw.w1p[5][0][0])[i] = (_Float16)((r < 32 && k < 4) ? P.r_embed_w[r * 4 + k] : 0.f);
    }
    __syncthreads();

    // ---- load input tile (12 features, zero-pad to 32) + guard pad ----
    for (int i = lane; i < 16 * 32; i += 32) {
        const int r = i >> 5, c = i & 31;
        const float v = P.x[(size_t)(row0 + r) * 12 + (c < 12 ? c : 0)];
        L.xbuf[r][c] = (c < 12) ? v : 0.f;
    }
    if (lane < 16) L.xpad[lane] = 0.f;

    // ---- 5 MLP branches: fire(0,s0) terrain(2,s1) burn(4,s2) soil(5,s3) storm(10,s5) ----
    const int offs[5]  = {0, 2, 4, 5, 10};
    const int slots[5] = {0, 1, 2, 3, 5};
#pragma unroll
    for (int g = 0; g < 5; ++g) {
        // A = shifted xbuf window; cols >= nf hit zero weights, row-15 overhang lands in zeroed xpad
        gemmL<true, true>(&L.hid[0][0], 64, &L.xbuf[0][offs[g]], 32, &sw.w1p[g][0][0], 32, P.mb1[g], 64, 32);
        gemmG<true, true>(&L.seqy[0][slots[g] * 32], 192, &L.hid[0][0], 64, P.mw2[g], P.mb2[g], 32, 64);
    }

    // ---- rain branch: embed + mamba(L=1) ----
    gemmL<true, false>(&L.gxl[0][0], 32, &L.xbuf[0][6], 32, &sw.w1p[5][0][0], 32, P.r_embed_b, 32, 32);
    gemmG<false, false>(&L.ring[0][0][0], 64, &L.gxl[0][0], 32, P.r_in_w,         nullptr, 64, 32); // xi
    gemmG<false, false>(&L.zb[0][0],      64, &L.gxl[0][0], 32, P.r_in_w + 64*32, nullptr, 64, 32); // z
    for (int i = lane; i < 16 * 64; i += 32) {
        const int r = i >> 6, d = i & 63;
        L.xc[r][d] = siluf(L.ring[0][r][d] * P.r_conv_w[d * 4 + 3] + P.r_conv_b[d]);
    }
    gemmG<false, false>(&L.dblv[0][0], 48, &L.xc[0][0], 64, P.r_xproj_w, nullptr, 34, 64);
    if (lane < 16) {
        float s = 0.f;
#pragma unroll
        for (int n = 0; n < 16; ++n) s += L.dblv[lane][2 + n] * L.dblv[lane][18 + n];
        L.bc[lane] = s;
    }
    for (int i = lane; i < 16 * 64; i += 32) {
        const int r = i >> 6, d = i & 63;
        const float dtv = softplusf(L.dblv[r][0] * P.r_dt_w[d * 2] +
                                    L.dblv[r][1] * P.r_dt_w[d * 2 + 1] + P.r_dt_b[d]);
        const float xv = L.xc[r][d];
        L.yb[r][d] = (dtv * xv * L.bc[r] + xv * P.r_D[d]) * siluf(L.zb[r][d]);
    }
    gemmG<false, false>(&L.seqy[0][4 * 32], 192, &L.yb[0][0], 64, P.r_out_w, nullptr, 32, 64);

    // ---- global embed + mamba over L=6 (streamed, all weights from staged f16 LDS) ----
    for (int i = lane; i < 16 * 64 * 16 / 4; i += 32)
        reinterpret_cast<float4*>(&L.h[0][0][0])[i] = float4{0.f, 0.f, 0.f, 0.f};

    for (int l = 0; l < 6; ++l) {
        gemmL<true,  false>(&L.gxl[0][0], 32, &L.seqy[0][l * 32], 192, &sw.gembed[0][0], 32, P.gembed_b, 32, 32);
        gemmL<false, false>(&L.ring[l & 3][0][0], 64, &L.gxl[0][0], 32, &sw.g_in[0][0],  32, nullptr, 64, 32);
        gemmL<false, false>(&L.zb[0][0],          64, &L.gxl[0][0], 32, &sw.g_in[64][0], 32, nullptr, 64, 32);
        // causal conv (k=4) + silu
        for (int i = lane; i < 16 * 64; i += 32) {
            const int r = i >> 6, d = i & 63;
            float acc = P.g_conv_b[d];
#pragma unroll
            for (int j = 0; j < 4; ++j) {
                const int ll = l - 3 + j;
                if (ll >= 0) acc += L.ring[ll & 3][r][d] * P.g_conv_w[d * 4 + j];
            }
            L.xc[r][d] = siluf(acc);
        }
        // xproj as N=48 (rows 34..47 zero weights -> zero outputs, no masking)
        gemmL<false, false>(&L.dblv[0][0], 48, &L.xc[0][0], 64, &sw.g_xproj[0][0], 64, nullptr, 48, 64);
        // selective scan step: 1024 (row,d) pairs over 32 lanes, vectorized state access
        for (int i = lane; i < 16 * 64; i += 32) {
            const int r = i >> 6, d = i & 63;
            const float dtv = softplusf(L.dblv[r][0] * P.g_dt_w[d * 2] +
                                        L.dblv[r][1] * P.g_dt_w[d * 2 + 1] + P.g_dt_b[d]);
            const float xv  = L.xc[r][d];
            const float dtx = dtv * xv;
            float hv[16], Bv[16], Cv[16];
#pragma unroll
            for (int t = 0; t < 4; ++t)
                reinterpret_cast<float4*>(hv)[t] = reinterpret_cast<const float4*>(&L.h[r][d][0])[t];
#pragma unroll
            for (int t = 0; t < 8; ++t) {
                reinterpret_cast<float2*>(Bv)[t] = *reinterpret_cast<const float2*>(&L.dblv[r][2 + 2*t]);
                reinterpret_cast<float2*>(Cv)[t] = *reinterpret_cast<const float2*>(&L.dblv[r][18 + 2*t]);
            }
            float y = 0.f;
#pragma unroll
            for (int n = 0; n < 16; ++n) {
                hv[n] = hv[n] * __expf(dtv * negA[d][n]) + dtx * Bv[n];
                y += hv[n] * Cv[n];
            }
#pragma unroll
            for (int t = 0; t < 4; ++t)
                reinterpret_cast<float4*>(&L.h[r][d][0])[t] = reinterpret_cast<const float4*>(hv)[t];
            y += xv * P.g_D[d];
            L.yb[r][d] = y * siluf(L.zb[r][d]);
        }
        gemmL<false, false>(&L.seqy[0][l * 32], 192, &L.yb[0][0], 64, &sw.g_out[0][0], 64, nullptr, 32, 64);
    }

    // ---- head ----
    gemmG<true, true>(&L.hid[0][0], 64, &L.seqy[0][0], 192, P.head_w1, P.head_b1, 64, 192);
    if (lane < 16) {
        float s = P.head_b2[0];
#pragma unroll
        for (int d = 0; d < 64; ++d) s += L.hid[lane][d] * P.head_w2[d];
        P.out[row0 + lane] = s;
    }
}

// Input leaf order: JAX pytree flattening (alphabetical dict keys, recursive).
enum {
    IN_BURN_B1 = 0, IN_BURN_B2, IN_BURN_W1, IN_BURN_W2,
    IN_FIRE_B1, IN_FIRE_B2, IN_FIRE_W1, IN_FIRE_W2,
    IN_GEMBED_B, IN_GEMBED_W,
    IN_GM_ALOG, IN_GM_D, IN_GM_CONV_B, IN_GM_CONV_W, IN_GM_DT_B, IN_GM_DT_W,
    IN_GM_IN_W, IN_GM_OUT_W, IN_GM_XPROJ_W,
    IN_HEAD_B1, IN_HEAD_B2, IN_HEAD_W1, IN_HEAD_W2,
    IN_RAIN_EMBED_B, IN_RAIN_EMBED_W,
    IN_RM_ALOG, IN_RM_D, IN_RM_CONV_B, IN_RM_CONV_W, IN_RM_DT_B, IN_RM_DT_W,
    IN_RM_IN_W, IN_RM_OUT_W, IN_RM_XPROJ_W,
    IN_SOIL_B1, IN_SOIL_B2, IN_SOIL_W1, IN_SOIL_W2,
    IN_STORM_B1, IN_STORM_B2, IN_STORM_W1, IN_STORM_W2,
    IN_TERRAIN_B1, IN_TERRAIN_B2, IN_TERRAIN_W1, IN_TERRAIN_W2,
    IN_X
};

extern "C" void kernel_launch(void* const* d_in, const int* in_sizes, int n_in,
                              void* d_out, int out_size, void* d_ws, size_t ws_size,
                              hipStream_t stream) {
    (void)n_in; (void)d_ws; (void)ws_size; (void)out_size;
    auto F = [&](int i) { return (const float*)d_in[i]; };

    KParams P;
    P.x = F(IN_X);
    const int w1i[5] = {IN_FIRE_W1, IN_TERRAIN_W1, IN_BURN_W1, IN_SOIL_W1, IN_STORM_W1};
    const int b1i[5] = {IN_FIRE_B1, IN_TERRAIN_B1, IN_BURN_B1, IN_SOIL_B1, IN_STORM_B1};
    const int w2i[5] = {IN_FIRE_W2, IN_TERRAIN_W2, IN_BURN_W2, IN_SOIL_W2, IN_STORM_W2};
    const int b2i[5] = {IN_FIRE_B2, IN_TERRAIN_B2, IN_BURN_B2, IN_SOIL_B2, IN_STORM_B2};
    for (int g = 0; g < 5; ++g) {
        P.mw1[g] = F(w1i[g]); P.mb1[g] = F(b1i[g]);
        P.mw2[g] = F(w2i[g]); P.mb2[g] = F(b2i[g]);
    }
    P.r_embed_w = F(IN_RAIN_EMBED_W); P.r_embed_b = F(IN_RAIN_EMBED_B);
    P.r_in_w = F(IN_RM_IN_W); P.r_conv_w = F(IN_RM_CONV_W); P.r_conv_b = F(IN_RM_CONV_B);
    P.r_xproj_w = F(IN_RM_XPROJ_W); P.r_dt_w = F(IN_RM_DT_W); P.r_dt_b = F(IN_RM_DT_B);
    P.r_D = F(IN_RM_D); P.r_out_w = F(IN_RM_OUT_W);
    P.gembed_w = F(IN_GEMBED_W); P.gembed_b = F(IN_GEMBED_B);
    P.g_in_w = F(IN_GM_IN_W); P.g_conv_w = F(IN_GM_CONV_W); P.g_conv_b = F(IN_GM_CONV_B);
    P.g_xproj_w = F(IN_GM_XPROJ_W); P.g_dt_w = F(IN_GM_DT_W); P.g_dt_b = F(IN_GM_DT_B);
    P.g_Alog = F(IN_GM_ALOG); P.g_D = F(IN_GM_D); P.g_out_w = F(IN_GM_OUT_W);
    P.head_w1 = F(IN_HEAD_W1); P.head_b1 = F(IN_HEAD_B1);
    P.head_w2 = F(IN_HEAD_W2); P.head_b2 = F(IN_HEAD_B2);
    P.out = (float*)d_out;

    const int rows   = in_sizes[IN_X] / 12;            // 32768
    const int blocks = rows / (ROWS * WAVES);          // 1024
    fused_multibranch_mamba<<<blocks, BLOCK, 0, stream>>>(P);
}
